// MLAAttention_30915174596911
// MI455X (gfx1250) — matrix operand order repaired and checked by reading, not compiled, not run on hardware
//
#include <hip/hip_runtime.h>
#include <hip/hip_bf16.h>
#include <stdint.h>

// ---------------------------------------------------------------------------
// MLA attention for MI455X (gfx1250): bf16 WMMA pipeline, f32 accumulate,
// async global->LDS staging (ASYNCcnt) with double-buffered GEMM tiles.
// ---------------------------------------------------------------------------

typedef __attribute__((ext_vector_type(16))) __bf16 v16bf;
typedef __attribute__((ext_vector_type(8)))  float  v8f;
typedef __hip_bfloat16 bf16_t;

#define NT     4096   // B*S tokens
#define SEQ    2048
#define DIM    2048
#define NH     16
#define QLORA  1536
#define KVLORA 512
#define ROPE_D 64
#define NOPE_D 64
#define V_D    128
#define QHD    128    // NOPE_D + ROPE_D
#define KVROW  576    // KVLORA + ROPE_D
#define KVBROW 3072   // NH*(NOPE_D+V_D)

#define WMMA_BF16(a, b, c) \
  __builtin_amdgcn_wmma_f32_16x16x32_bf16(false, (a), false, (b), (short)0, (c), false, false)

// --- CDNA5 async global->LDS copy (tracked by ASYNCcnt) --------------------
// dsaddr is wave-relative; a generic LDS pointer's low 32 bits are exactly
// that offset.
static __device__ inline uint32_t lds_off(const void* p) {
  return (uint32_t)(uintptr_t)p;
}
static __device__ inline void async_ld_b128(uint32_t ldsdst, const void* gbase, int32_t voff) {
  asm volatile("global_load_async_to_lds_b128 %0, %1, %2"
               :: "v"(ldsdst), "v"(voff), "s"(gbase) : "memory");
}
static __device__ inline void wait_async0() {
  asm volatile("s_wait_asynccnt 0" ::: "memory");
}

// A fragment (16x32 bf16, M x K): lane<16 -> K 0..15; lane>=16 -> K 16..31.
static __device__ inline v16bf frag_ld_a(const bf16_t* base, int stride) {
  int lane = threadIdx.x & 31;
  const bf16_t* p = base + (lane & 15) * stride + (lane >> 4) * 16;
  union { v16bf v; uint4 q[2]; } u;
  u.q[0] = *(const uint4*)(p);
  u.q[1] = *(const uint4*)(p + 8);
  return u.v;
}

// B fragment (32x16 bf16, K x N) from a transposed (N x K row-major) tile:
// lane = column, half-wave selects K group; 16 contiguous K per lane.
static __device__ inline v16bf frag_ld_b(const bf16_t* baseT, int stride) {
  int lane = threadIdx.x & 31;
  const bf16_t* p = baseT + (lane & 15) * stride + (lane >> 4) * 16;
  union { v16bf v; uint4 q[2]; } u;
  u.q[0] = *(const uint4*)(p);
  u.q[1] = *(const uint4*)(p + 8);
  return u.v;
}

static __device__ inline void store_out(float* p, float x)  { *p = x; }
static __device__ inline void store_out(bf16_t* p, float x) { *p = __float2bfloat16(x); }

// ---------------------------------------------------------------------------
// Conversion kernels
// ---------------------------------------------------------------------------
__global__ void cvt_f32_bf16(const float* __restrict__ in, bf16_t* __restrict__ out, size_t n) {
  size_t i = (size_t)blockIdx.x * 256 + threadIdx.x;
  if (i < n) out[i] = __float2bfloat16(in[i]);
}

// in: K x N (row-major f32) -> out: N x K (row-major bf16)
__global__ void transpose_f32_bf16(const float* __restrict__ in, bf16_t* __restrict__ out,
                                   int K, int N) {
  int n = blockIdx.x * 16 + threadIdx.x;
  int k = blockIdx.y * 16 + threadIdx.y;
  if (n < N && k < K) out[(size_t)n * K + k] = __float2bfloat16(in[(size_t)k * N + n]);
}

// ---------------------------------------------------------------------------
// Tiled bf16 WMMA GEMM: C[M,N] = A[M,K] (lda) * Bt[N,K]^T
// 128x128 block tile, K-step 32, double-buffered LDS tiles filled with
// async-to-LDS b128s; 8 waves each computing a 32x64 tile (2x4 WMMA tiles).
// ---------------------------------------------------------------------------
template <typename OutT>
__global__ __launch_bounds__(256) void gemm_bf16(
    const bf16_t* __restrict__ A, int lda,
    const bf16_t* __restrict__ Bt,
    OutT* __restrict__ C, int ldc,
    int M, int N, int K)
{
  __shared__ bf16_t As[2][128][40];   // 80 B row stride (16B aligned)
  __shared__ bf16_t Bs[2][128][40];   // Bs[buf][n][k]

  const int tid  = threadIdx.x;
  const int wave = tid >> 5, lane = tid & 31;
  const int wm = wave & 3, wn = wave >> 2;
  const int hi = lane >> 4, lcol = lane & 15;
  const size_t m0 = (size_t)blockIdx.y * 128;
  const int    n0 = blockIdx.x * 128;

  v8f acc[2][4];
  for (int i = 0; i < 2; i++)
    for (int j = 0; j < 4; j++)
      acc[i][j] = (v8f){0.f,0.f,0.f,0.f,0.f,0.f,0.f,0.f};

  // async-stage one 128x32 K-tile of A and Bt into buffer bb
  auto stage = [&](int k0, int bb) {
    for (int i = 0; i < 2; i++) {
      int idx = tid + i * 256;          // b128 index over 128x32 (512 total)
      int row = idx >> 2;
      int c8  = (idx & 3) * 8;
      async_ld_b128(lds_off(&As[bb][row][c8]), A,
                    (int32_t)(((m0 + row) * lda + k0 + c8) * 2));
      int n = n0 + row;
      int nn = (n < N) ? n : 0;         // OOB rows load garbage; store is guarded
      async_ld_b128(lds_off(&Bs[bb][row][c8]), Bt,
                    (int32_t)(((size_t)nn * K + k0 + c8) * 2));
    }
  };

  const int nk = K >> 5;
  stage(0, 0);
  for (int t = 0; t < nk; t++) {
    wait_async0();        // own async fills done
    __syncthreads();      // everyone's fills done / previous reads retired
    if (t + 1 < nk) stage((t + 1) << 5, (t + 1) & 1);
    const int bb = t & 1;

    v16bf af[2], bfr[4];
    for (int mt = 0; mt < 2; mt++) af[mt]  = frag_ld_a(&As[bb][wm * 32 + mt * 16][0], 40);
    for (int nt = 0; nt < 4; nt++) bfr[nt] = frag_ld_b(&Bs[bb][wn * 64 + nt * 16][0], 40);
    for (int mt = 0; mt < 2; mt++)
      for (int nt = 0; nt < 4; nt++)
        acc[mt][nt] = WMMA_BF16(af[mt], bfr[nt], acc[mt][nt]);
  }

  for (int mt = 0; mt < 2; mt++)
    for (int nt = 0; nt < 4; nt++)
      for (int v = 0; v < 8; v++) {
        int row = (int)m0 + wm * 32 + mt * 16 + v + hi * 8;
        int col = n0 + wn * 64 + nt * 16 + lcol;
        if (col < N) store_out(&C[(size_t)row * ldc + col], acc[mt][nt][v]);
      }
}

// ---------------------------------------------------------------------------
// RoPE (interleaved even/odd pairs), in-place on bf16 buffers
// ---------------------------------------------------------------------------
__global__ void rope_q(bf16_t* __restrict__ q) {
  int idx = blockIdx.x * 256 + threadIdx.x;         // NT*NH*32 pairs
  if (idx >= NT * NH * 32) return;
  int j = idx & 31;
  int h = (idx >> 5) & (NH - 1);
  int t = idx >> 9;
  int pos = t & (SEQ - 1);
  float ang = (float)pos * __powf(10000.0f, -(float)j / 32.0f);
  float s, c; __sincosf(ang, &s, &c);
  size_t base = (size_t)t * DIM + h * QHD + NOPE_D + 2 * j;
  float e = __bfloat162float(q[base]), o = __bfloat162float(q[base + 1]);
  q[base]     = __float2bfloat16(e * c - o * s);
  q[base + 1] = __float2bfloat16(o * c + e * s);
}

__global__ void rope_k(bf16_t* __restrict__ kvc) {
  int idx = blockIdx.x * 256 + threadIdx.x;         // NT*32 pairs
  if (idx >= NT * 32) return;
  int j = idx & 31;
  int t = idx >> 5;
  int pos = t & (SEQ - 1);
  float ang = (float)pos * __powf(10000.0f, -(float)j / 32.0f);
  float s, c; __sincosf(ang, &s, &c);
  size_t base = (size_t)t * KVROW + KVLORA + 2 * j;
  float e = __bfloat162float(kvc[base]), o = __bfloat162float(kvc[base + 1]);
  kvc[base]     = __float2bfloat16(e * c - o * s);
  kvc[base + 1] = __float2bfloat16(o * c + e * s);
}

// ---------------------------------------------------------------------------
// Flash attention: 128 queries/block, 64-key steps, causal, online softmax.
// grid = (SEQ/128, B*NH). 8 waves; wave w owns query rows 16w..16w+15.
// ---------------------------------------------------------------------------
#define QSTR 136
#define KSTR 136
#define VSTR 72
#define PSTR 72

__global__ __launch_bounds__(256) void mla_attention(
    const bf16_t* __restrict__ q,    // [NT][DIM]   head h at cols h*QHD (pe roped)
    const bf16_t* __restrict__ kv,   // [NT][KVBROW] head h: k_nope +0..63, v +64..191
    const bf16_t* __restrict__ kvc,  // [NT][KVROW]  roped k_pe at cols KVLORA..
    bf16_t* __restrict__ o)          // [NT][NH*V_D]
{
  __shared__ alignas(16) char smem[(64 * KSTR + 128 * VSTR + 8 * 16 * PSTR) * 2]; // 54272 B
  static_assert(128 * QSTR * 2 <= (64 * KSTR + 128 * VSTR + 8 * 16 * PSTR) * 2, "smem");

  bf16_t* qs = (bf16_t*)smem;                                   // phase 0 only
  bf16_t* ks = (bf16_t*)smem;                                   // [64][KSTR]
  bf16_t* vs = (bf16_t*)(smem + 64 * KSTR * 2);                 // Vt [128][VSTR]
  bf16_t* ps = (bf16_t*)(smem + (64 * KSTR + 128 * VSTR) * 2);  // P per wave [16][PSTR]

  const int tid = threadIdx.x, wave = tid >> 5, lane = tid & 31;
  const int hi = lane >> 4, lcol = lane & 15;
  const int bh = blockIdx.y;
  const int b = bh >> 4, h = bh & 15;
  const int q0 = blockIdx.x * 128;
  const int tq0 = b * SEQ + q0;

  // --- async-stage Q tile (128 x 128), then hoist fragments into registers ---
  for (int i = 0; i < 8; i++) {
    int idx = tid + i * 256;            // b128 over 128x128 (2048 total)
    int row = idx >> 4;
    int c8  = (idx & 15) * 8;
    async_ld_b128(lds_off(&qs[row * QSTR + c8]), q,
                  (int32_t)(((size_t)(tq0 + row) * DIM + h * QHD + c8) * 2));
  }
  wait_async0();
  __syncthreads();
  v16bf qf[4];
  for (int kt = 0; kt < 4; kt++)
    qf[kt] = frag_ld_a(qs + wave * 16 * QSTR + kt * 32, QSTR);
  // Q-fragment DS reads must land before the region is recycled for K/V
  asm volatile("s_wait_dscnt 0" ::: "memory");

  v8f oacc[8];
  for (int t = 0; t < 8; t++) oacc[t] = (v8f){0.f,0.f,0.f,0.f,0.f,0.f,0.f,0.f};
  float mrow[8], lrow[8];
  for (int v = 0; v < 8; v++) { mrow[v] = -1e30f; lrow[v] = 0.f; }
  const float scale = 0.08838834764831845f;  // 1/sqrt(128)

  for (int k0 = 0; k0 < q0 + 128; k0 += 64) {
    __syncthreads();   // previous iteration's K/V reads are retired
    // async-stage K tile [64 keys][128 d]: d<64 k_nope from kv, d>=64 k_pe from kvc
    for (int i = 0; i < 4; i++) {
      int idx = tid + i * 256;          // b128 over 64x128 (1024 total)
      int row = idx >> 4;
      int c8  = (idx & 15) * 8;
      int tok = b * SEQ + k0 + row;
      if (c8 < 64)
        async_ld_b128(lds_off(&ks[row * KSTR + c8]), kv,
                      (int32_t)(((size_t)tok * KVBROW + h * (NOPE_D + V_D) + c8) * 2));
      else
        async_ld_b128(lds_off(&ks[row * KSTR + c8]), kvc,
                      (int32_t)(((size_t)tok * KVROW + KVLORA + (c8 - 64)) * 2));
    }
    // stage V^T tile [128 d][64 keys] (transpose: vectorized reads, scalar DS stores)
    for (int i = 0; i < 8; i++) {
      int idx = tid + i * 256;          // vec4 over 64x128 (2048 total)
      int key = idx >> 5;
      int d4  = (idx & 31) * 4;
      int tok = b * SEQ + k0 + key;
      uint2 vv = *(const uint2*)(kv + (size_t)tok * KVBROW + h * (NOPE_D + V_D) + NOPE_D + d4);
      bf16_t tmp[4];
      *(uint2*)tmp = vv;
      for (int j = 0; j < 4; j++) vs[(d4 + j) * VSTR + key] = tmp[j];
    }
    wait_async0();
    __syncthreads();

    // scores S = Q Kt : wave rows 16w..16w+15, all 64 keys (4 N-tiles x 4 K-steps)
    v8f sacc[4];
    for (int nt = 0; nt < 4; nt++) sacc[nt] = (v8f){0.f,0.f,0.f,0.f,0.f,0.f,0.f,0.f};
    for (int kt = 0; kt < 4; kt++)
      for (int nt = 0; nt < 4; nt++) {
        v16bf kf = frag_ld_b(ks + nt * 16 * KSTR + kt * 32, KSTR);
        sacc[nt] = WMMA_BF16(qf[kt], kf, sacc[nt]);
      }

    // online softmax in C-layout: element v -> row v + 8*hi, lane -> column lcol
    float nmax[8];
    for (int v = 0; v < 8; v++) nmax[v] = mrow[v];
    for (int nt = 0; nt < 4; nt++)
      for (int v = 0; v < 8; v++) {
        float x = sacc[nt][v] * scale;
        int key = k0 + nt * 16 + lcol;
        int qi  = q0 + wave * 16 + v + hi * 8;
        x = (key <= qi) ? x : -1e30f;
        sacc[nt][v] = x;
        nmax[v] = fmaxf(nmax[v], x);
      }
    for (int off = 1; off < 16; off <<= 1)
      for (int v = 0; v < 8; v++)
        nmax[v] = fmaxf(nmax[v], __shfl_xor(nmax[v], off, 32));

    float corr[8], rsum[8];
    for (int v = 0; v < 8; v++) {
      corr[v] = __expf(mrow[v] - nmax[v]);
      mrow[v] = nmax[v];
      rsum[v] = 0.f;
    }
    for (int nt = 0; nt < 4; nt++)
      for (int v = 0; v < 8; v++) {
        float p = __expf(sacc[nt][v] - mrow[v]);
        sacc[nt][v] = p;
        rsum[v] += p;
      }
    for (int off = 1; off < 16; off <<= 1)
      for (int v = 0; v < 8; v++)
        rsum[v] += __shfl_xor(rsum[v], off, 32);
    for (int v = 0; v < 8; v++) lrow[v] = lrow[v] * corr[v] + rsum[v];
    for (int t = 0; t < 8; t++)
      for (int v = 0; v < 8; v++) oacc[t][v] *= corr[v];

    // P (16x64) -> per-wave LDS, reload as A fragments for O += P V
    bf16_t* pw = ps + wave * 16 * PSTR;
    for (int nt = 0; nt < 4; nt++)
      for (int v = 0; v < 8; v++)
        pw[(v + hi * 8) * PSTR + nt * 16 + lcol] = __float2bfloat16(sacc[nt][v]);
    v16bf pf[2];
    pf[0] = frag_ld_a(pw, PSTR);
    pf[1] = frag_ld_a(pw + 32, PSTR);
    for (int t = 0; t < 8; t++)
      for (int kk = 0; kk < 2; kk++) {
        v16bf vf = frag_ld_b(vs + t * 16 * VSTR + kk * 32, VSTR);
        oacc[t] = WMMA_BF16(pf[kk], vf, oacc[t]);
      }
  }

  for (int v = 0; v < 8; v++) lrow[v] = 1.0f / lrow[v];
  for (int t = 0; t < 8; t++)
    for (int v = 0; v < 8; v++) {
      int row = q0 + wave * 16 + v + hi * 8;
      int tok = b * SEQ + row;
      int col = h * V_D + t * 16 + lcol;
      o[(size_t)tok * (NH * V_D) + col] = __float2bfloat16(oacc[t][v] * lrow[v]);
    }
}

// ---------------------------------------------------------------------------
// Host orchestration
// ---------------------------------------------------------------------------
extern "C" void kernel_launch(void* const* d_in, const int* in_sizes, int n_in,
                              void* d_out, int out_size, void* d_ws, size_t ws_size,
                              hipStream_t stream) {
  const float* hs   = (const float*)d_in[0];
  const float* Wqa  = (const float*)d_in[1];
  const float* Wqb  = (const float*)d_in[2];
  const float* Wkva = (const float*)d_in[3];
  const float* Wkvb = (const float*)d_in[4];
  const float* Wo   = (const float*)d_in[5];
  float* out = (float*)d_out;

  bf16_t* w = (bf16_t*)d_ws;
  size_t off = 0;
  auto take = [&](size_t n) { bf16_t* p = w + off; off += n; return p; };

  bf16_t* hs_b  = take((size_t)NT * DIM);
  bf16_t* WqaT  = take((size_t)QLORA * DIM);
  bf16_t* WqbT  = take((size_t)(NH * QHD) * QLORA);
  bf16_t* WkvaT = take((size_t)KVROW * DIM);
  bf16_t* WkvbT = take((size_t)KVBROW * KVLORA);
  bf16_t* WoT   = take((size_t)DIM * (NH * V_D));
  bf16_t* qlat  = take((size_t)NT * QLORA);
  bf16_t* qbuf  = take((size_t)NT * (NH * QHD));
  bf16_t* kvc   = take((size_t)NT * KVROW);
  bf16_t* kvb   = take((size_t)NT * KVBROW);
  bf16_t* attn  = hs_b;  // hs_b dead after the two projections from it

  // conversions
  size_t nhs = (size_t)NT * DIM;
  cvt_f32_bf16<<<(unsigned)((nhs + 255) / 256), 256, 0, stream>>>(hs, hs_b, nhs);
  dim3 tb(16, 16);
  transpose_f32_bf16<<<dim3(QLORA / 16, DIM / 16), tb, 0, stream>>>(Wqa, WqaT, DIM, QLORA);
  transpose_f32_bf16<<<dim3((NH * QHD) / 16, QLORA / 16), tb, 0, stream>>>(Wqb, WqbT, QLORA, NH * QHD);
  transpose_f32_bf16<<<dim3(KVROW / 16, DIM / 16), tb, 0, stream>>>(Wkva, WkvaT, DIM, KVROW);
  transpose_f32_bf16<<<dim3(KVBROW / 16, KVLORA / 16), tb, 0, stream>>>(Wkvb, WkvbT, KVLORA, KVBROW);
  transpose_f32_bf16<<<dim3(DIM / 16, (NH * V_D) / 16), tb, 0, stream>>>(Wo, WoT, NH * V_D, DIM);

  // projections
  gemm_bf16<bf16_t><<<dim3(QLORA / 128, NT / 128), 256, 0, stream>>>(
      hs_b, DIM, WqaT, qlat, QLORA, NT, QLORA, DIM);
  gemm_bf16<bf16_t><<<dim3((NH * QHD) / 128, NT / 128), 256, 0, stream>>>(
      qlat, QLORA, WqbT, qbuf, NH * QHD, NT, NH * QHD, QLORA);
  gemm_bf16<bf16_t><<<dim3((KVROW + 127) / 128, NT / 128), 256, 0, stream>>>(
      hs_b, DIM, WkvaT, kvc, KVROW, NT, KVROW, DIM);
  gemm_bf16<bf16_t><<<dim3(KVBROW / 128, NT / 128), 256, 0, stream>>>(
      kvc, KVROW, WkvbT, kvb, KVBROW, NT, KVBROW, KVLORA);

  // rope
  rope_q<<<(NT * NH * 32 + 255) / 256, 256, 0, stream>>>(qbuf);
  rope_k<<<(NT * 32 + 255) / 256, 256, 0, stream>>>(kvc);

  // attention
  mla_attention<<<dim3(SEQ / 128, 2 * NH), 256, 0, stream>>>(qbuf, kvb, kvc, attn);

  // output projection (f32 out)
  gemm_bf16<float><<<dim3(DIM / 128, NT / 128), 256, 0, stream>>>(
      attn, NH * V_D, WoT, out, DIM, NT, DIM, NH * V_D);
}
